// GRACE_50105088475331
// MI455X (gfx1250) — compile-verified, matrix operand order checked by reading, not run on hardware
//
#include <hip/hip_runtime.h>
#include <hip/hip_bf16.h>
#include <math.h>

// ---------------------------------------------------------------------------
// GRACE forward for MI455X (gfx1250, wave32).
//  - GEMMs: v_wmma_f32_16x16x32_bf16, K fully unrolled (templated), weights
//    pre-packed to bf16 in WMMA fragment order (contiguous 32B per lane).
//  - Row-clamped A loads: no divergent predication in the WMMA loop.
//  - Self-loop term fused into the GEMM epilogue.
//  - Scatter: wave-per-edge float4 gathers + global_atomic_add_f32 into
//    L2-resident accumulators (working set < 192 MB L2).
// ---------------------------------------------------------------------------

typedef __attribute__((ext_vector_type(16))) __bf16 v16bf;
typedef __attribute__((ext_vector_type(8)))  float  v8f;

// ---- pack B[K,N] f32 (row-major) -> bf16 fragments ------------------------
// Packed index: (((nt*KB + kb)*32 + lane)*16 + j) where lane's element j maps
// to k_local = 16*(j>>3) + 8*(lane>>4) + (j&7), n = nt*16 + (lane&15).
__global__ void pack_b_bf16(const float* __restrict__ B, __bf16* __restrict__ Bp,
                            int K, int N) {
  const int idx = blockIdx.x * blockDim.x + threadIdx.x;
  if (idx >= K * N) return;
  const int KB    = K >> 5;
  const int j     = idx & 15;
  const int lane  = (idx >> 4) & 31;
  const int blkid = idx >> 9;            // nt*KB + kb
  const int kb    = blkid % KB;
  const int nt    = blkid / KB;
  const int k     = kb * 32 + ((j >> 3) << 4) + ((lane >> 4) << 3) + (j & 7);
  const int n     = nt * 16 + (lane & 15);
  Bp[idx] = (__bf16)B[k * N + n];
}

// ---- WMMA GEMM: D = act(A @ B (+bias)); optional Dself = D * dinv[row]^2 --
template <int K>
__global__ __launch_bounds__(128) void gemm_bf16_wmma(
    const float* __restrict__ A, const __bf16* __restrict__ Bp,
    const float* __restrict__ bias, float* __restrict__ D,
    const float* __restrict__ dinv, float* __restrict__ Dself,
    int M, int N, int do_relu) {
  constexpr int KB = K / 32;
  const int wave  = threadIdx.x >> 5;
  const int lane  = threadIdx.x & 31;
  const int m0    = blockIdx.x * 16;
  const int nt    = blockIdx.y * 4 + wave;
  const int n0    = nt * 16;
  if (n0 >= N) return;                        // wave-uniform (EXEC stays full)

  const int mr    = lane & 15;
  const int khalf = lane >> 4;
  const int m     = min(m0 + mr, M - 1);      // clamp: garbage rows never stored
  const int n     = n0 + mr;

  const float*  ar = A + (long long)m * K + 8 * khalf;
  const __bf16* br = Bp + ((long long)nt * KB * 32 + lane) * 16;

  v8f acc = {};
#pragma unroll
  for (int kbi = 0; kbi < KB; ++kbi) {
    const int kb = kbi * 32;
    const float4 a0 = *(const float4*)(ar + kb);
    const float4 a1 = *(const float4*)(ar + kb + 4);
    const float4 a2 = *(const float4*)(ar + kb + 16);
    const float4 a3 = *(const float4*)(ar + kb + 20);
    v16bf a;
    a[0]=(__bf16)a0.x; a[1]=(__bf16)a0.y; a[2]=(__bf16)a0.z; a[3]=(__bf16)a0.w;
    a[4]=(__bf16)a1.x; a[5]=(__bf16)a1.y; a[6]=(__bf16)a1.z; a[7]=(__bf16)a1.w;
    a[8]=(__bf16)a2.x; a[9]=(__bf16)a2.y; a[10]=(__bf16)a2.z; a[11]=(__bf16)a2.w;
    a[12]=(__bf16)a3.x; a[13]=(__bf16)a3.y; a[14]=(__bf16)a3.z; a[15]=(__bf16)a3.w;
    const v16bf b = *(const v16bf*)(br + (long long)kbi * 32 * 16);
    acc = __builtin_amdgcn_wmma_f32_16x16x32_bf16(
        false, a, false, b, (short)0, acc, false, false);
  }

  const float bv = bias ? bias[n] : 0.0f;
#pragma unroll
  for (int v = 0; v < 8; ++v) {
    const int row = m0 + v + 8 * khalf;       // C/D striping: VGPR v -> M = v (+8)
    if (row < M) {
      float val = acc[v] + bv;
      if (do_relu) val = fmaxf(val, 0.0f);
      const long long o = (long long)row * N + n;
      D[o] = val;
      if (Dself) { const float dd = dinv[row]; Dself[o] = val * dd * dd; }
    }
  }
}

// ---------------- degree / norm helpers ------------------------------------
__global__ void fill_f32(float* __restrict__ p, float v, int n) {
  int i = blockIdx.x * blockDim.x + threadIdx.x;
  if (i < n) p[i] = v;
}

__global__ void deg_accum(const int* __restrict__ col, float* __restrict__ deg, int E) {
  int e = blockIdx.x * blockDim.x + threadIdx.x;
  if (e < E) atomicAdd(&deg[col[e]], 1.0f);
}

__global__ void rsqrt_inplace(float* __restrict__ p, int n) {
  int i = blockIdx.x * blockDim.x + threadIdx.x;
  if (i < n) p[i] = __frsqrt_rn(p[i]);
}

// ---- out[col[e],:] += H[row[e],:] * dinv[row]*dinv[col] -------------------
// F=128: wave = 1 edge (32 float4 chunks). F=64: wave = 2 edges (16 chunks).
template <int F>
__global__ void edge_scatter(const int* __restrict__ row, const int* __restrict__ col,
                             const float* __restrict__ dinv, const float* __restrict__ H,
                             float* __restrict__ out, int E) {
  constexpr int CH  = F / 4;
  constexpr int EPW = 32 / CH;
  const int lane = threadIdx.x & 31;
  const int sub  = lane / CH;
  const int q    = lane % CH;
  const long long wid0   = ((long long)blockIdx.x * blockDim.x + threadIdx.x) >> 5;
  const long long nwaves = ((long long)gridDim.x * blockDim.x) >> 5;
  for (long long wv = wid0; wv * EPW < E; wv += nwaves) {
    const long long e = wv * EPW + sub;
    if (e >= E) continue;
    const int r = row[e], c = col[e];
    const float w = dinv[r] * dinv[c];
    const float4 h = *(const float4*)(H + (long long)r * F + q * 4);
    float* o = out + (long long)c * F + q * 4;
    atomicAdd(o + 0, h.x * w);
    atomicAdd(o + 1, h.y * w);
    atomicAdd(o + 2, h.z * w);
    atomicAdd(o + 3, h.w * w);
  }
}

// x = relu(x + b[f])
__global__ void bias_relu(float* __restrict__ x, const float* __restrict__ b,
                          long long total, int fmask) {
  long long i = (long long)blockIdx.x * blockDim.x + threadIdx.x;
  if (i < total) x[i] = fmaxf(x[i] + b[(int)(i & fmask)], 0.0f);
}

// L2-normalize rows of U[N,128] into out (one wave32 per node)
__global__ void l2norm_rows128(const float* __restrict__ U, float* __restrict__ out, int N) {
  const int node = blockIdx.x * (blockDim.x >> 5) + (threadIdx.x >> 5);
  const int lane = threadIdx.x & 31;
  if (node >= N) return;
  const float* u = U + (long long)node * 128;
  float vals[4], s = 0.0f;
#pragma unroll
  for (int j = 0; j < 4; ++j) { vals[j] = u[lane + 32 * j]; s += vals[j] * vals[j]; }
#pragma unroll
  for (int off = 16; off > 0; off >>= 1) s += __shfl_xor(s, off, 32);
  const float inv = 1.0f / fmaxf(sqrtf(s), 1e-12f);
  float* o = out + (long long)node * 128;
#pragma unroll
  for (int j = 0; j < 4; ++j) o[lane + 32 * j] = vals[j] * inv;
}

// c = softmax(T[node,:] @ Wc2[64,3] + bc2)
__global__ void cluster_head(const float* __restrict__ T, const float* __restrict__ Wc2,
                             const float* __restrict__ bc2, float* __restrict__ out, int N) {
  const int node = blockIdx.x * blockDim.x + threadIdx.x;
  if (node >= N) return;
  float l0 = bc2[0], l1 = bc2[1], l2 = bc2[2];
  const float* t = T + (long long)node * 64;
#pragma unroll 8
  for (int c = 0; c < 64; ++c) {
    const float tv = t[c];
    l0 += tv * Wc2[c * 3 + 0];
    l1 += tv * Wc2[c * 3 + 1];
    l2 += tv * Wc2[c * 3 + 2];
  }
  const float mx = fmaxf(l0, fmaxf(l1, l2));
  const float e0 = __expf(l0 - mx), e1 = __expf(l1 - mx), e2 = __expf(l2 - mx);
  const float inv = 1.0f / (e0 + e1 + e2);
  float* o = out + (long long)node * 3;
  o[0] = e0 * inv; o[1] = e1 * inv; o[2] = e2 * inv;
}

// ---------------------------------------------------------------------------
static inline int cdiv(long long a, long long b) { return (int)((a + b - 1) / b); }

extern "C" void kernel_launch(void* const* d_in, const int* in_sizes, int n_in,
                              void* d_out, int out_size, void* d_ws, size_t ws_size,
                              hipStream_t stream) {
  const float* x[2]  = { (const float*)d_in[0], (const float*)d_in[2] };
  const int*   ei[2] = { (const int*)d_in[1],   (const int*)d_in[3]   };
  const float* W1  = (const float*)d_in[4];
  const float* b1  = (const float*)d_in[5];
  const float* W2  = (const float*)d_in[6];
  const float* b2  = (const float*)d_in[7];
  const float* Wp1 = (const float*)d_in[8];  const float* bp1 = (const float*)d_in[9];
  const float* Wp2 = (const float*)d_in[10]; const float* bp2 = (const float*)d_in[11];
  const float* Wc1 = (const float*)d_in[12]; const float* bc1 = (const float*)d_in[13];
  const float* Wc2 = (const float*)d_in[14]; const float* bc2 = (const float*)d_in[15];

  const int N = in_sizes[0] / 128;   // 50000
  const int E = in_sizes[1] / 2;     // 800000

  // ---- workspace layout ----
  float* w      = (float*)d_ws;
  float* dinv   = w;                                   // N
  float* bufH   = dinv + N;                            // N*128
  float* bufAgg = bufH + (long long)N * 128;           // N*128 (aggregator / hidden)
  float* bufT   = bufAgg + (long long)N * 128;         // N*64  (layer-2 aggregate / z)
  __bf16* bwp   = (__bf16*)(bufT + (long long)N * 64); // packed bf16 weights
  __bf16* W1p  = bwp;                 // 128*128
  __bf16* W2p  = W1p + 128 * 128;     // 128*64
  __bf16* Wp1p = W2p + 128 * 64;      // 64*64
  __bf16* Wp2p = Wp1p + 64 * 64;      // 64*128
  __bf16* Wc1p = Wp2p + 64 * 128;     // 64*64

  float* out  = (float*)d_out;
  float* zout[2] = { out, out + (long long)N * 128 };
  float* cbase   = out + (long long)2 * N * 128;
  float* cout[2] = { cbase, cbase + (long long)N * 3 };

  const dim3 blk256(256);
  const int tilesM = cdiv(N, 16);

  // ---- pack weights to bf16 fragment order (tiny, once per call) ----
  pack_b_bf16<<<cdiv(128 * 128, 256), blk256, 0, stream>>>(W1,  W1p,  128, 128);
  pack_b_bf16<<<cdiv(128 * 64, 256),  blk256, 0, stream>>>(W2,  W2p,  128, 64);
  pack_b_bf16<<<cdiv(64 * 64, 256),   blk256, 0, stream>>>(Wp1, Wp1p, 64, 64);
  pack_b_bf16<<<cdiv(64 * 128, 256),  blk256, 0, stream>>>(Wp2, Wp2p, 64, 128);
  pack_b_bf16<<<cdiv(64 * 64, 256),   blk256, 0, stream>>>(Wc1, Wc1p, 64, 64);

  for (int g = 0; g < 2; ++g) {
    const int* rows = ei[g];
    const int* cols = ei[g] + E;

    // --- symmetric normalization: dinv = rsqrt(1 + indegree) ---
    fill_f32<<<cdiv(N, 256), blk256, 0, stream>>>(dinv, 1.0f, N);
    deg_accum<<<cdiv(E, 256), blk256, 0, stream>>>(cols, dinv, E);
    rsqrt_inplace<<<cdiv(N, 256), blk256, 0, stream>>>(dinv, N);

    // --- GCN layer 1: H = x@W1 (+fused self-loop init) ; scatter ; relu+b1 ---
    gemm_bf16_wmma<128><<<dim3(tilesM, 2), dim3(128), 0, stream>>>(
        x[g], W1p, nullptr, bufH, dinv, bufAgg, N, 128, 0);
    edge_scatter<128><<<cdiv((long long)E * 32, 256), blk256, 0, stream>>>(
        rows, cols, dinv, bufH, bufAgg, E);
    bias_relu<<<cdiv((long long)N * 128, 256), blk256, 0, stream>>>(
        bufAgg, b1, (long long)N * 128, 127);

    // --- GCN layer 2: H = agg@W2 (+self-loop) ; scatter ; relu+b2 -> z ---
    gemm_bf16_wmma<128><<<dim3(tilesM, 1), dim3(128), 0, stream>>>(
        bufAgg, W2p, nullptr, bufH, dinv, bufT /*N*64 agg*/, N, 64, 0);
    edge_scatter<64><<<cdiv((long long)E * 16, 256), blk256, 0, stream>>>(
        rows, cols, dinv, bufH, bufT, E);
    bias_relu<<<cdiv((long long)N * 64, 256), blk256, 0, stream>>>(
        bufT, b2, (long long)N * 64, 63);
    // z lives in bufT[N*64]; move role: use bufAgg (N*64 slice) as hidden now
    float* z = bufT;
    float* hid = bufAgg;

    // --- instance head: l2norm(relu(z@Wp1+bp1) @ Wp2 + bp2) ---
    gemm_bf16_wmma<64><<<dim3(tilesM, 1), dim3(128), 0, stream>>>(
        z, Wp1p, bp1, hid, nullptr, nullptr, N, 64, 1);
    gemm_bf16_wmma<64><<<dim3(tilesM, 2), dim3(128), 0, stream>>>(
        hid, Wp2p, bp2, bufH, nullptr, nullptr, N, 128, 0);
    l2norm_rows128<<<cdiv(N, 8), blk256, 0, stream>>>(bufH, zout[g], N);

    // --- cluster head: softmax(relu(z@Wc1+bc1) @ Wc2 + bc2) ---
    gemm_bf16_wmma<64><<<dim3(tilesM, 1), dim3(128), 0, stream>>>(
        z, Wc1p, bc1, hid, nullptr, nullptr, N, 64, 1);
    cluster_head<<<cdiv(N, 256), blk256, 0, stream>>>(hid, Wc2, bc2, cout[g], N);
  }
}
